// DRAW_62122406969969
// MI455X (gfx1250) — matrix operand order repaired and checked by reading, not compiled, not run on hardware
//
#include <hip/hip_runtime.h>

// ---------------------------------------------------------------------------
// DRAW recurrent VAE on MI455X (gfx1250).
// Big LSTM/projection GEMMs -> v_wmma_f32_16x16x32_f16 (f16 in, f32 acc),
// guard-free inner loop (zero-padded weights/activations), fragment-ordered
// LDS tiles so fragment reads are ds_load_b128. Tiny attention einsums on
// VALU with LDS staging. Canvas lives in d_out across all 16 steps (L2).
// ---------------------------------------------------------------------------

typedef __attribute__((ext_vector_type(16))) _Float16 v16h;
typedef __attribute__((ext_vector_type(8)))  _Float16 v8h;
typedef __attribute__((ext_vector_type(4)))  _Float16 v4h;
typedef __attribute__((ext_vector_type(8)))  float    v8f;

#define T_STEPS 16
#define BATCH   4096
#define AA      64
#define BB      64
#define NN      12
#define ZS      100
#define ENC     400
#define DEC     400

#define LDH     416    // padded hidden stride (400 -> 416, mult of 32)
#define LDZ     128    // padded latent stride (100 -> 128)

// ----------------------------- small utility kernels -----------------------

__global__ void fill0_kernel(float* __restrict__ p, int n) {
    int i = blockIdx.x * blockDim.x + threadIdx.x;
    if (i < n) p[i] = 0.f;
}

// dst is (Npad x Kp) f16, zero padded from src (N x K) f32.
__global__ void cvt_pad_f16_kernel(const float* __restrict__ src,
                                   _Float16* __restrict__ dst,
                                   int N, int K, int Kp, int total) {
    int i = blockIdx.x * blockDim.x + threadIdx.x;
    if (i >= total) return;
    int n = i / Kp, k = i - n * Kp;
    float v = (n < N && k < K) ? src[(size_t)n * K + k] : 0.f;
    dst[i] = (_Float16)v;
}

__global__ void addvec_kernel(const float* __restrict__ a,
                              const float* __restrict__ b,
                              float* __restrict__ o, int n) {
    int i = blockIdx.x * blockDim.x + threadIdx.x;
    if (i < n) o[i] = a[i] + b[i];
}

// z[b, j] = mu + exp(logsig) * eps  (mu/sig/z use LDZ stride, eps is dense ZS)
__global__ void z_kernel(const float* __restrict__ mu,
                         const float* __restrict__ logsig,
                         const float* __restrict__ eps_t,
                         float* __restrict__ z, int n) {
    int i = blockIdx.x * blockDim.x + threadIdx.x;
    if (i >= n) return;
    int b = i / ZS, j = i - b * ZS;
    int p = b * LDZ + j;
    z[p] = mu[p] + __expf(logsig[p]) * eps_t[i];
}

// PyTorch gate order i,f,g,o contiguous along the 4*H axis; h/c use ldh stride.
__global__ void lstm_kernel(const float* __restrict__ gates,
                            float* __restrict__ cstate,
                            float* __restrict__ h, int H, int ldh, int total) {
    int idx = blockIdx.x * blockDim.x + threadIdx.x;
    if (idx >= total) return;
    int b = idx / H, j = idx - b * H;
    const float* g = gates + (size_t)b * 4 * H;
    float gi = g[j], gf = g[H + j], gg = g[2 * H + j], go = g[3 * H + j];
    float si = 1.f / (1.f + __expf(-gi));
    float sf = 1.f / (1.f + __expf(-gf));
    float so = 1.f / (1.f + __expf(-go));
    int p = b * ldh + j;
    float c2 = sf * cstate[p] + si * tanhf(gg);
    cstate[p] = c2;
    h[p] = so * tanhf(c2);
}

// ----------------------------- WMMA GEMM -----------------------------------
// O[M,N] (+)= X_f32[M,Kp] * W_f16[Npad,Kp]^T (+ bias[N] on init).
// Kp multiple of 32 (zero padded), W zero padded to Npad = gridDim.x*64 rows.
// Block: 128 thr = 4 waves; block tile 32(M) x 64(N); 2 WMMA per K-chunk.
// LDS tiles are stored in WMMA fragment order -> v16h reads = 2x ds_load_b128.
__global__ __launch_bounds__(128)
void gemm_wmma_kernel(const float* __restrict__ X, int ldx,
                      const _Float16* __restrict__ W, int ldw,
                      const float* __restrict__ bias,
                      float* __restrict__ O, int ldo,
                      int Ntot, int Kp, int accum)
{
    __shared__ __align__(32) _Float16 sAf[2][32][16];  // [m-tile][lane][j]
    __shared__ __align__(32) _Float16 sBf[4][32][16];  // [wave][lane][j]

    const int tid  = threadIdx.x;
    const int lane = tid & 31;
    const int wave = tid >> 5;
    const int half = lane >> 4;
    const int l16  = lane & 15;
    const int m0   = blockIdx.y * 32;
    const int n0   = blockIdx.x * 64;
    const int ncol = n0 + wave * 16 + l16;

    // A staging coordinates for this thread: row 0..15, 4 consecutive k.
    const int arow = tid >> 3;
    const int acol = (tid & 7) * 4;
    const int lA   = arow + 16 * ((acol >> 3) & 1);   // fragment lane
    const int jA   = (acol & 7) + 8 * (acol >> 4);    // fragment slot

    v8f acc0, acc1;
    if (accum) {
#pragma unroll
        for (int r = 0; r < 8; ++r) {
            int m = r + 8 * half;
            acc0[r] = (ncol < Ntot) ? O[(size_t)(m0 + m) * ldo + ncol] : 0.f;
            acc1[r] = (ncol < Ntot) ? O[(size_t)(m0 + 16 + m) * ldo + ncol] : 0.f;
        }
    } else {
        float bv = (bias && ncol < Ntot) ? bias[ncol] : 0.f;
#pragma unroll
        for (int r = 0; r < 8; ++r) { acc0[r] = bv; acc1[r] = bv; }
    }

    for (int kb = 0; kb < Kp; kb += 32) {
        // Stage A: two 16x32 f32 tiles; float4 load -> 4-half LDS store.
#pragma unroll
        for (int mt = 0; mt < 2; ++mt) {
            const float4 v =
                *(const float4*)&X[(size_t)(m0 + mt * 16 + arow) * ldx + kb + acol];
            v4h hv;
            hv[0] = (_Float16)v.x; hv[1] = (_Float16)v.y;
            hv[2] = (_Float16)v.z; hv[3] = (_Float16)v.w;
            *(v4h*)&sAf[mt][lA][jA] = hv;
        }
        // Stage B: 64x32 halfs; 16-byte load -> 16-byte LDS store, no guards.
#pragma unroll
        for (int i = 0; i < 2; ++i) {
            int g  = tid + 128 * i;          // 0..255 groups
            int n  = g >> 2;                 // 0..63
            int kg = g & 3;                  // 8-half group within 32
            const v8h w8 = *(const v8h*)&W[(size_t)(n0 + n) * ldw + kb + kg * 8];
            *(v8h*)&sBf[n >> 4][(n & 15) + 16 * (kg & 1)][(kg >> 1) * 8] = w8;
        }
        __syncthreads();

        const v16h b  = *(const v16h*)sBf[wave][lane];
        const v16h a0 = *(const v16h*)sAf[0][lane];
        const v16h a1 = *(const v16h*)sAf[1][lane];
        acc0 = __builtin_amdgcn_wmma_f32_16x16x32_f16(
                   false, a0, false, b, (short)0, acc0, false, false);
        acc1 = __builtin_amdgcn_wmma_f32_16x16x32_f16(
                   false, a1, false, b, (short)0, acc1, false, false);
        __syncthreads();
    }

    if (ncol < Ntot) {
#pragma unroll
        for (int r = 0; r < 8; ++r) {
            int m = r + 8 * half;
            O[(size_t)(m0 + m) * ldo + ncol]      = acc0[r];
            O[(size_t)(m0 + 16 + m) * ldo + ncol] = acc1[r];
        }
    }
}

// ----------------------------- attention filterbank ------------------------
__global__ __launch_bounds__(128)
void attn_kernel(const float* __restrict__ h_dec, int ldh,
                 const float* __restrict__ w_attn,
                 const float* __restrict__ b_attn,
                 float* __restrict__ Fx, float* __restrict__ Fy,
                 float* __restrict__ gam)
{
    const int b = blockIdx.x;
    const int tid = threadIdx.x;
    __shared__ float red[5][128];
    __shared__ float p[5];
    __shared__ float F[2][NN][64];
    __shared__ float rowsum[2][NN];

#pragma unroll
    for (int i = 0; i < 5; ++i) {
        float s = 0.f;
        for (int k = tid; k < DEC; k += 128)
            s += h_dec[(size_t)b * ldh + k] * w_attn[i * DEC + k];
        red[i][tid] = s;
    }
    __syncthreads();
    for (int off = 64; off > 0; off >>= 1) {
        if (tid < off)
#pragma unroll
            for (int i = 0; i < 5; ++i) red[i][tid] += red[i][tid + off];
        __syncthreads();
    }
    if (tid < 5) p[tid] = red[tid][0] + b_attn[tid];
    __syncthreads();

    const float gx     = 0.5f * (AA + 1) * (p[0] + 1.f);
    const float gy     = 0.5f * (BB + 1) * (p[1] + 1.f);
    const float sigma2 = __expf(p[2]);
    const float delta  = ((float)(AA - 1) / (float)(NN - 1)) * __expf(p[3]);
    const float gamma  = __expf(p[4]);
    const float hcen   = (float)NN / 2.f + 0.5f;

    for (int e = tid; e < 2 * NN * 64; e += 128) {
        int which = e / (NN * 64);
        int n = (e % (NN * 64)) >> 6;
        int a = e & 63;
        float g  = which ? gy : gx;
        float mu = ((float)n - hcen) * delta + g;
        float t  = ((float)a - mu) / (2.f * sigma2);
        F[which][n][a] = __expf(-t * t);
    }
    __syncthreads();
    if (tid < 2 * NN) {
        int which = tid / NN, n = tid % NN;
        float s = 0.f;
        for (int a = 0; a < 64; ++a) s += F[which][n][a];
        rowsum[which][n] = s + 1e-9f;
    }
    __syncthreads();
    for (int e = tid; e < 2 * NN * 64; e += 128) {
        int which = e / (NN * 64);
        int n = (e % (NN * 64)) >> 6;
        int a = e & 63;
        float v = F[which][n][a] / rowsum[which][n];
        if (which) Fy[(size_t)b * NN * 64 + n * 64 + a] = v;
        else       Fx[(size_t)b * NN * 64 + n * 64 + a] = v;
    }
    if (tid == 0) gam[b] = gamma;
}

// ----------------------------- read (glimpse) ------------------------------
__global__ __launch_bounds__(256)
void glimpse_kernel(const float* __restrict__ x, const float* __restrict__ c,
                    const float* __restrict__ Fx, const float* __restrict__ Fy,
                    const float* __restrict__ gam, float* __restrict__ r)
{
    const int b = blockIdx.x;
    const int tid = threadIdx.x;
    __shared__ float img[2][BB][AA];
    __shared__ float sFx[NN][64], sFy[NN][64];
    __shared__ float tmp[2][NN][64];

    for (int e = tid; e < BB * AA; e += 256) {
        float xv = x[(size_t)b * AA * BB + e];
        float cv = c[(size_t)b * AA * BB + e];
        img[0][e >> 6][e & 63] = xv;
        img[1][e >> 6][e & 63] = xv - 1.f / (1.f + __expf(-cv));
    }
    for (int e = tid; e < NN * 64; e += 256) {
        sFx[e >> 6][e & 63] = Fx[(size_t)b * NN * 64 + e];
        sFy[e >> 6][e & 63] = Fy[(size_t)b * NN * 64 + e];
    }
    __syncthreads();

    for (int e = tid; e < 2 * NN * 64; e += 256) {
        int w = e / (NN * 64);
        int n = (e % (NN * 64)) >> 6;
        int a = e & 63;
        float s = 0.f;
        for (int Bi = 0; Bi < BB; ++Bi) s += sFy[n][Bi] * img[w][Bi][a];
        tmp[w][n][a] = s;
    }
    __syncthreads();

    const float gamma = gam[b];
    for (int e = tid; e < 2 * NN * NN; e += 256) {
        int w = e / (NN * NN);
        int n = (e % (NN * NN)) / NN;
        int m = e % NN;
        float s = 0.f;
        for (int a = 0; a < 64; ++a) s += tmp[w][n][a] * sFx[m][a];
        r[(size_t)b * 2 * NN * NN + w * NN * NN + n * NN + m] = s * gamma;
    }
}

// ----------------------------- write back to canvas ------------------------
__global__ __launch_bounds__(256)
void write_kernel(const float* __restrict__ wimg,
                  const float* __restrict__ Fx, const float* __restrict__ Fy,
                  const float* __restrict__ gam, float* __restrict__ c)
{
    const int b = blockIdx.x;
    const int tid = threadIdx.x;
    __shared__ float sFx[NN][64], sFy[NN][64], sw[NN][NN], tmp[NN][64];

    for (int e = tid; e < NN * 64; e += 256) {
        sFx[e >> 6][e & 63] = Fx[(size_t)b * NN * 64 + e];
        sFy[e >> 6][e & 63] = Fy[(size_t)b * NN * 64 + e];
    }
    if (tid < NN * NN) sw[tid / NN][tid % NN] = wimg[(size_t)b * NN * NN + tid];
    __syncthreads();

    for (int e = tid; e < NN * 64; e += 256) {
        int n = e >> 6, a = e & 63;
        float s = 0.f;
#pragma unroll
        for (int m = 0; m < NN; ++m) s += sw[n][m] * sFx[m][a];
        tmp[n][a] = s;
    }
    __syncthreads();

    const float inv_g = 1.f / gam[b];
    for (int e = tid; e < BB * AA; e += 256) {
        int Bi = e >> 6, Ai = e & 63;
        float s = 0.f;
#pragma unroll
        for (int n = 0; n < NN; ++n) s += sFy[n][Bi] * tmp[n][Ai];
        c[(size_t)b * AA * BB + e] += s * inv_g;
    }
}

// ----------------------------- host-side orchestration ---------------------

static inline void launch_gemm(const float* X, int ldx, const _Float16* W, int ldw,
                               const float* bias, float* O, int ldo,
                               int M, int N, int Kp, int accum, hipStream_t s)
{
    dim3 grid((N + 63) / 64, M / 32);
    gemm_wmma_kernel<<<grid, 128, 0, s>>>(X, ldx, W, ldw, bias, O, ldo, N, Kp, accum);
}

extern "C" void kernel_launch(void* const* d_in, const int* in_sizes, int n_in,
                              void* d_out, int out_size, void* d_ws, size_t ws_size,
                              hipStream_t stream)
{
    (void)in_sizes; (void)n_in; (void)out_size; (void)ws_size;

    const float* x        = (const float*)d_in[0];
    const float* eps      = (const float*)d_in[1];
    const float* w_enc_ih = (const float*)d_in[2];
    const float* w_enc_hh = (const float*)d_in[3];
    const float* b_enc_ih = (const float*)d_in[4];
    const float* b_enc_hh = (const float*)d_in[5];
    const float* w_dec_ih = (const float*)d_in[6];
    const float* w_dec_hh = (const float*)d_in[7];
    const float* b_dec_ih = (const float*)d_in[8];
    const float* b_dec_hh = (const float*)d_in[9];
    const float* w_mu     = (const float*)d_in[10];
    const float* b_mu     = (const float*)d_in[11];
    const float* w_sig    = (const float*)d_in[12];
    const float* b_sig    = (const float*)d_in[13];
    const float* w_attn   = (const float*)d_in[14];
    const float* b_attn   = (const float*)d_in[15];
    const float* w_write  = (const float*)d_in[16];
    const float* b_write  = (const float*)d_in[17];

    float* canvas = (float*)d_out;   // c lives in d_out across all steps

    const int KIN  = 2 * NN * NN + DEC;  // 688
    const int KINP = 704;                // 688 -> 704 (mult 32)

    // ---- workspace layout (floats first, then f16 weights) ----
    float* f = (float*)d_ws;
    size_t off = 0;
    auto take = [&](size_t n) { float* p = f + off; off += n; return p; };

    float* h_enc = take((size_t)BATCH * LDH);
    float* s_enc = take((size_t)BATCH * LDH);
    float* h_dec = take((size_t)BATCH * LDH);
    float* s_dec = take((size_t)BATCH * LDH);
    float* r     = take((size_t)BATCH * 2 * NN * NN);   // ld 288 (mult 32)
    float* gates = take((size_t)BATCH * 4 * ENC);       // ld 1600
    float* mu    = take((size_t)BATCH * LDZ);
    float* sig   = take((size_t)BATCH * LDZ);
    float* z     = take((size_t)BATCH * LDZ);
    float* gam   = take((size_t)BATCH);
    float* wimg  = take((size_t)BATCH * NN * NN);       // ld 144
    float* b_enc_c = take(4 * ENC);
    float* b_dec_c = take(4 * DEC);

    // Fx/Fy alias the gates buffer: lifetimes never overlap with gates.
    float* Fx = gates;
    float* Fy = gates + (size_t)BATCH * NN * 64;

    _Float16* hreg = (_Float16*)(f + off);
    size_t hoff = 0;
    auto takeh = [&](size_t n) { _Float16* p = hreg + hoff; hoff += n; return p; };
    _Float16* w16_enc_ih = takeh((size_t)1600 * KINP);  // 1600 x 704
    _Float16* w16_enc_hh = takeh((size_t)1600 * LDH);   // 1600 x 416
    _Float16* w16_dec_ih = takeh((size_t)1600 * LDZ);   // 1600 x 128
    _Float16* w16_dec_hh = takeh((size_t)1600 * LDH);   // 1600 x 416
    _Float16* w16_mu     = takeh((size_t)128 * LDH);    // 128(pad) x 416
    _Float16* w16_sig    = takeh((size_t)128 * LDH);
    _Float16* w16_write  = takeh((size_t)192 * LDH);    // 192(pad) x 416

    // ---- weight conversion to padded f16 (once per launch) ----
    auto cvt = [&](const float* s, _Float16* d, int N, int K, int Kp, int Npad) {
        int total = Npad * Kp;
        cvt_pad_f16_kernel<<<(total + 255) / 256, 256, 0, stream>>>(s, d, N, K, Kp, total);
    };
    cvt(w_enc_ih, w16_enc_ih, 1600, KIN, KINP, 1600);
    cvt(w_enc_hh, w16_enc_hh, 1600, ENC, LDH, 1600);
    cvt(w_dec_ih, w16_dec_ih, 1600, ZS,  LDZ, 1600);
    cvt(w_dec_hh, w16_dec_hh, 1600, DEC, LDH, 1600);
    cvt(w_mu,     w16_mu,     ZS,   ENC, LDH, 128);
    cvt(w_sig,    w16_sig,    ZS,   ENC, LDH, 128);
    cvt(w_write,  w16_write,  NN * NN, DEC, LDH, 192);

    addvec_kernel<<<(4 * ENC + 255) / 256, 256, 0, stream>>>(b_enc_ih, b_enc_hh, b_enc_c, 4 * ENC);
    addvec_kernel<<<(4 * DEC + 255) / 256, 256, 0, stream>>>(b_dec_ih, b_dec_hh, b_dec_c, 4 * DEC);

    auto zero = [&](float* p, size_t n) {
        fill0_kernel<<<(int)((n + 255) / 256), 256, 0, stream>>>(p, (int)n);
    };
    zero(canvas, (size_t)BATCH * AA * BB);
    zero(h_enc, (size_t)BATCH * LDH);   // zeroed pads stay zero forever
    zero(s_enc, (size_t)BATCH * LDH);
    zero(h_dec, (size_t)BATCH * LDH);
    zero(s_dec, (size_t)BATCH * LDH);
    zero(z,     (size_t)BATCH * LDZ);
    zero(mu,    (size_t)BATCH * LDZ);
    zero(sig,   (size_t)BATCH * LDZ);

    // ---- 16 sequential DRAW steps ----
    for (int t = 0; t < T_STEPS; ++t) {
        attn_kernel<<<BATCH, 128, 0, stream>>>(h_dec, LDH, w_attn, b_attn, Fx, Fy, gam);
        glimpse_kernel<<<BATCH, 256, 0, stream>>>(x, canvas, Fx, Fy, gam, r);

        // encoder gates = [r, h_dec] @ Wi^T + b + h_enc @ Wh^T
        launch_gemm(r,     2 * NN * NN, w16_enc_ih,               KINP, b_enc_c,
                    gates, 4 * ENC, BATCH, 4 * ENC, 2 * NN * NN, 0, stream);
        launch_gemm(h_dec, LDH,         w16_enc_ih + 2 * NN * NN, KINP, nullptr,
                    gates, 4 * ENC, BATCH, 4 * ENC, LDH, 1, stream);
        launch_gemm(h_enc, LDH,         w16_enc_hh,               LDH,  nullptr,
                    gates, 4 * ENC, BATCH, 4 * ENC, LDH, 1, stream);
        lstm_kernel<<<(BATCH * ENC + 255) / 256, 256, 0, stream>>>(
            gates, s_enc, h_enc, ENC, LDH, BATCH * ENC);

        // latent
        launch_gemm(h_enc, LDH, w16_mu,  LDH, b_mu,  mu,  LDZ, BATCH, ZS, LDH, 0, stream);
        launch_gemm(h_enc, LDH, w16_sig, LDH, b_sig, sig, LDZ, BATCH, ZS, LDH, 0, stream);
        z_kernel<<<(BATCH * ZS + 255) / 256, 256, 0, stream>>>(
            mu, sig, eps + (size_t)t * BATCH * ZS, z, BATCH * ZS);

        // decoder gates = z @ Wi^T + b + h_dec @ Wh^T
        launch_gemm(z,     LDZ, w16_dec_ih, LDZ, b_dec_c,
                    gates, 4 * DEC, BATCH, 4 * DEC, LDZ, 0, stream);
        launch_gemm(h_dec, LDH, w16_dec_hh, LDH, nullptr,
                    gates, 4 * DEC, BATCH, 4 * DEC, LDH, 1, stream);
        lstm_kernel<<<(BATCH * DEC + 255) / 256, 256, 0, stream>>>(
            gates, s_dec, h_dec, DEC, LDH, BATCH * DEC);

        // second attention + write patch + canvas update
        attn_kernel<<<BATCH, 128, 0, stream>>>(h_dec, LDH, w_attn, b_attn, Fx, Fy, gam);
        launch_gemm(h_dec, LDH, w16_write, LDH, b_write, wimg, NN * NN,
                    BATCH, NN * NN, LDH, 0, stream);
        write_kernel<<<BATCH, 256, 0, stream>>>(wimg, Fx, Fy, gam, canvas);
    }
}
